// Quantize_42434276885049
// MI455X (gfx1250) — matrix operand order repaired and checked by reading, not compiled, not run on hardware
//
#include <hip/hip_runtime.h>
#include <hip/hip_bf16.h>

typedef __attribute__((ext_vector_type(16))) __bf16 v16bf;
typedef __attribute__((ext_vector_type(8)))  float  v8f;

#define DIM     256
#define NEMBED  4096
#define NROWS   32768                  // 8*64*64
#define QELEMS  (NROWS * DIM)          // 8388608
#define DIFF_OFF QELEMS
#define IND_OFF  (QELEMS + 1)

// ---------------- Kernel 1: pack codebook -------------------------------
// embed is [DIM][NEMBED] row-major.  Produce:
//   eT   : float  [NEMBED][DIM]  (for exact fp32 gather of winning codes)
//   eHi  : bf16   [NEMBED][DIM]  (high part)
//   eLo  : bf16   [NEMBED][DIM]  (residual)
//   enorm: float  [NEMBED]       (||e||^2)
__global__ void vq_pack(const float* __restrict__ embed,
                        float* __restrict__ eT,
                        __bf16* __restrict__ eHi,
                        __bf16* __restrict__ eLo,
                        float* __restrict__ enorm) {
    const int e = blockIdx.x * blockDim.x + threadIdx.x;
    if (e >= NEMBED) return;
    float s = 0.f;
#pragma unroll 4
    for (int d = 0; d < DIM; ++d) {
        float v = embed[(size_t)d * NEMBED + e];   // coalesced across threads
        s += v * v;
        eT[(size_t)e * DIM + d] = v;
        __bf16 h = (__bf16)v;
        eHi[(size_t)e * DIM + d] = h;
        eLo[(size_t)e * DIM + d] = (__bf16)(v - (float)h);
    }
    enorm[e] = s;
}

// ------- helpers: 16x256 bf16 hi/lo tile, global<->regs<->LDS ----------
__device__ __forceinline__ void load_tile(const __bf16* __restrict__ eHi,
                                          const __bf16* __restrict__ eLo,
                                          int colBase, int colLd, int ksLd,
                                          uint4 r[4]) {
    const uint4* gH = (const uint4*)(eHi + (size_t)(colBase + colLd) * DIM + ksLd);
    const uint4* gL = (const uint4*)(eLo + (size_t)(colBase + colLd) * DIM + ksLd);
    r[0] = gH[0]; r[1] = gH[1];
    r[2] = gL[0]; r[3] = gL[1];
}

__device__ __forceinline__ void store_tile(__bf16 (*sH)[264], __bf16 (*sL)[264],
                                           int colLd, int ksLd, const uint4 r[4]) {
    uint4* dH = (uint4*)(&sH[colLd][ksLd]);
    uint4* dL = (uint4*)(&sL[colLd][ksLd]);
    dH[0] = r[0]; dH[1] = r[1];
    dL[0] = r[2]; dL[1] = r[3];
}

// ---------------- Kernel 2: distance GEMM + argmin + gather -------------
__global__ __launch_bounds__(256)
void vq_main(const float* __restrict__ input,
             const __bf16* __restrict__ eHi,
             const __bf16* __restrict__ eLo,
             const float* __restrict__ eT,
             const float* __restrict__ enorm,
             float* __restrict__ out,
             float* __restrict__ parts) {
    // Double-buffered B tile: 16 cols x 256 K, padded stride 264 bf16
    // (=132 dwords) so lane n's b128 read hits banks 4n..4n+3.
    __shared__ __bf16 sHi[2][16][264];
    __shared__ __bf16 sLo[2][16][264];
    __shared__ int    sIdx[8][16];

    const int tid  = threadIdx.x;
    const int lane = tid & 31;
    const int wv   = tid >> 5;          // wave 0..7, owns 16 rows
    const int m    = lane & 15;
    const int hi16 = lane >> 4;         // 0 or 1 (lane half)
    const int rowBase = blockIdx.x * 128 + wv * 16;

    // ---- preload A fragments (register-resident bf16 hi/lo split) ----
    // 16-bit A 16x32 layout: lanes 0-15 row M=lane, K {0..7,16..23};
    // lanes 16-31 row M=lane-16, K {8..15,24..31}.
    v16bf aHi[8], aLo[8];
    {
        const float* xr = input + (size_t)(rowBase + m) * DIM;
        const int kHalf = hi16 * 8;
#pragma unroll
        for (int s = 0; s < 8; ++s) {
            const int kb = s * 32 + kHalf;
            float f[16];
            float4 t0 = *(const float4*)(xr + kb);
            float4 t1 = *(const float4*)(xr + kb + 4);
            float4 t2 = *(const float4*)(xr + kb + 16);
            float4 t3 = *(const float4*)(xr + kb + 20);
            f[0]=t0.x;  f[1]=t0.y;  f[2]=t0.z;  f[3]=t0.w;
            f[4]=t1.x;  f[5]=t1.y;  f[6]=t1.z;  f[7]=t1.w;
            f[8]=t2.x;  f[9]=t2.y;  f[10]=t2.z; f[11]=t2.w;
            f[12]=t3.x; f[13]=t3.y; f[14]=t3.z; f[15]=t3.w;
#pragma unroll
            for (int j = 0; j < 16; ++j) {
                __bf16 h = (__bf16)f[j];
                aHi[s][j] = h;
                aLo[s][j] = (__bf16)(f[j] - (float)h);
            }
        }
    }

    float minv[8];
    int   mini[8];
#pragma unroll
    for (int r = 0; r < 8; ++r) { minv[r] = 3.4e38f; mini[r] = 0; }

    const int colLd = tid >> 4;          // col this thread stages (0..15)
    const int ksLd  = (tid & 15) * 16;   // 16 bf16 (32B) per thread per matrix

    // ---- software pipeline: prefetch chunk 0 ----
    uint4 pf[4];
    load_tile(eHi, eLo, 0, colLd, ksLd, pf);
    store_tile(sHi[0], sLo[0], colLd, ksLd, pf);
    __syncthreads();

    for (int chunk = 0; chunk < NEMBED / 16; ++chunk) {
        const int buf = chunk & 1;
        // issue next tile's global loads; latency hidden under the WMMAs
        int nc = chunk + 1; if (nc > NEMBED / 16 - 1) nc = NEMBED / 16 - 1;
        load_tile(eHi, eLo, nc * 16, colLd, ksLd, pf);

        const int colBase = chunk * 16;
        const float en = enorm[colBase + m];

        // three independent accumulator chains (hi*hi, hi*lo, lo*hi)
        v8f a0 = {0.f,0.f,0.f,0.f,0.f,0.f,0.f,0.f};
        v8f a1 = {0.f,0.f,0.f,0.f,0.f,0.f,0.f,0.f};
        v8f a2 = {0.f,0.f,0.f,0.f,0.f,0.f,0.f,0.f};
#pragma unroll
        for (int s = 0; s < 8; ++s) {
            // B 32x16 layout: lane n holds col N=n&15; lanes 0-15 K {0..7,16..23},
            // lanes 16-31 K {8..15,24..31} of this K-step (matches A).
            const int ko = s * 32 + hi16 * 8;
            union { uint4 u[2]; v16bf v; } bh, bl;
            const uint4* pH = (const uint4*)(&sHi[buf][m][ko]);
            const uint4* pL = (const uint4*)(&sLo[buf][m][ko]);
            bh.u[0] = pH[0]; bh.u[1] = pH[2];
            bl.u[0] = pL[0]; bl.u[1] = pL[2];
            a0 = __builtin_amdgcn_wmma_f32_16x16x32_bf16(false, aHi[s], false, bh.v,
                                                         (short)0, a0, false, false);
            a1 = __builtin_amdgcn_wmma_f32_16x16x32_bf16(false, aHi[s], false, bl.v,
                                                         (short)0, a1, false, false);
            a2 = __builtin_amdgcn_wmma_f32_16x16x32_bf16(false, aLo[s], false, bh.v,
                                                         (short)0, a2, false, false);
        }

        // stage the prefetched tile into the other buffer, one barrier/iter
        store_tile(sHi[buf ^ 1], sLo[buf ^ 1], colLd, ksLd, pf);
        __syncthreads();

        // dist (minus the per-row constant ||x||^2) = ||e||^2 - 2 x.e
        const int col = colBase + m;
#pragma unroll
        for (int r = 0; r < 8; ++r) {
            float dv = en - 2.f * (a0[r] + a1[r] + a2[r]);
            if (dv < minv[r]) { minv[r] = dv; mini[r] = col; }
        }
    }

    // ---- cross-lane argmin within each 16-lane half (xor<16 stays in half)
#pragma unroll
    for (int off = 1; off < 16; off <<= 1) {
#pragma unroll
        for (int r = 0; r < 8; ++r) {
            float ov = __shfl_xor(minv[r], off, 32);
            int   oi = __shfl_xor(mini[r], off, 32);
            if (ov < minv[r] || (ov == minv[r] && oi < mini[r])) {
                minv[r] = ov; mini[r] = oi;
            }
        }
    }
    // C layout: lanes 0-15 VGPR r <-> M=r ; lanes 16-31 VGPR r <-> M=8+r
    if (lane == 0) {
#pragma unroll
        for (int r = 0; r < 8; ++r) {
            sIdx[wv][r] = mini[r];
            out[IND_OFF + rowBase + r] = (float)mini[r];
        }
    } else if (lane == 16) {
#pragma unroll
        for (int r = 0; r < 8; ++r) {
            sIdx[wv][8 + r] = mini[r];
            out[IND_OFF + rowBase + 8 + r] = (float)mini[r];
        }
    }
    __syncthreads();

    // ---- gather winning codes (contiguous fp32 rows), straight-through
    //      output x + (q - x) (matches JAX rounding), diff partial sum.
    float local = 0.f;
    for (int r = 0; r < 16; ++r) {
        const int row = rowBase + r;
        const int idx = sIdx[wv][r];
        const float* q = eT    + (size_t)idx * DIM;
        const float* x = input + (size_t)row * DIM;
#pragma unroll
        for (int i = 0; i < 8; ++i) {
            const int d = lane + i * 32;
            float qv = q[d];
            float xv = x[d];
            float t  = qv - xv;
            out[(size_t)row * DIM + d] = xv + t;
            local += t * t;
        }
    }
#pragma unroll
    for (int off = 1; off < 32; off <<= 1)
        local += __shfl_xor(local, off, 32);
    if (lane == 0)
        parts[blockIdx.x * 8 + wv] = local;
}

// ---------------- Kernel 3: deterministic diff reduction ----------------
__global__ void vq_reduce(const float* __restrict__ parts, float* __restrict__ out) {
    __shared__ float sm[256];
    float s = 0.f;
#pragma unroll
    for (int i = 0; i < 8; ++i) s += parts[threadIdx.x + i * 256];  // fixed order
    sm[threadIdx.x] = s;
    __syncthreads();
    for (int off = 128; off > 0; off >>= 1) {
        if (threadIdx.x < off) sm[threadIdx.x] += sm[threadIdx.x + off];
        __syncthreads();
    }
    if (threadIdx.x == 0) out[DIFF_OFF] = sm[0] * (1.0f / (float)QELEMS);
}

extern "C" void kernel_launch(void* const* d_in, const int* in_sizes, int n_in,
                              void* d_out, int out_size, void* d_ws, size_t ws_size,
                              hipStream_t stream) {
    const float* input = (const float*)d_in[0];   // [8,64,64,256] fp32
    const float* embed = (const float*)d_in[1];   // [256,4096]   fp32
    float* out = (float*)d_out;                   // quantize | diff | indices

    char* ws = (char*)d_ws;
    float*  eT    = (float*)(ws);                 // 4,194,304 B
    __bf16* eHi   = (__bf16*)(ws + 4194304);      // 2,097,152 B
    __bf16* eLo   = (__bf16*)(ws + 6291456);      // 2,097,152 B
    float*  enormp= (float*)(ws + 8388608);       //    16,384 B
    float*  parts = (float*)(ws + 8404992);       //     8,192 B (256 blocks * 8 waves)

    vq_pack  <<<NEMBED / 256, 256, 0, stream>>>(embed, eT, eHi, eLo, enormp);
    vq_main  <<<NROWS / 128,  256, 0, stream>>>(input, eHi, eLo, eT, enormp, out, parts);
    vq_reduce<<<1,            256, 0, stream>>>(parts, out);
}
